// LSTM_84576495993102
// MI455X (gfx1250) — matrix operand order repaired and checked by reading, not compiled, not run on hardware
//
#include <hip/hip_runtime.h>

typedef __attribute__((ext_vector_type(16))) _Float16 v16h;
typedef __attribute__((ext_vector_type(8)))  _Float16 v8h;
typedef __attribute__((ext_vector_type(8)))  float    v8f;
typedef __attribute__((ext_vector_type(4)))  int      v4i;

#define H_   128     // hidden size
#define G4_  512     // 4*H gate rows
#define T_   2048    // sequence length
#define B_   128     // batch
#define BT_  16      // batch rows per workgroup (one WMMA M-tile)
#define LDK_ 136     // padded LDS row stride in halves (128+8) -> conflict-free
#define CHX_ 64      // layer-0 x chunk length (timesteps)
#define LDX0_ 40     // layer-0 chunk row stride in halves (32+8) -> conflict-free

// CDNA5 async global->LDS path (ASYNCcnt); signature probed in round 2/3.
#if __has_builtin(__builtin_amdgcn_global_load_async_to_lds_b128) && \
    __has_builtin(__builtin_amdgcn_s_wait_asynccnt)
#define USE_ASYNC_LDS 1
#define GAS_ __attribute__((address_space(1)))
#define LAS_ __attribute__((address_space(3)))
#endif

// Hardware v_rcp_f32 instead of IEEE-precise division (which lowers to a
// ~10-instruction div_scale/rcp/fma/div_fmas chain on the serial critical path).
__device__ __forceinline__ float fast_rcp(float x) { return __builtin_amdgcn_rcpf(x); }
__device__ __forceinline__ float sigmoid_f(float x) { return fast_rcp(1.0f + __expf(-x)); }
__device__ __forceinline__ float tanh_f(float x) {
    return 1.0f - 2.0f * fast_rcp(1.0f + __expf(2.0f * x));
}

__device__ __forceinline__ v16h cat8(v8h a, v8h b) {
    return __builtin_shufflevector(a, b, 0,1,2,3,4,5,6,7,8,9,10,11,12,13,14,15);
}

// One LSTM layer, persistent over T; each workgroup owns BT_ batch rows.
// ALL weight B-fragments live in registers (W_hh: 128 VGPRs, W_ih: 32*KXS VGPRs),
// loaded once straight from HBM with per-lane fp32->fp16 conversion — per-wave
// rows are disjoint, so no LDS staging is needed at all.
// KXS     = 32-wide K-steps for the x path (1 for layer0, 4 for layers 1/2)
// CHUNK_X = layer 0: stage CHX_ timesteps of fp32 x into LDS at once
// KIN     = real input width (6 or 128); zero-padded to KXS*32 in fragments
template<int KXS, bool CHUNK_X, int KIN>
__launch_bounds__(256)
__global__ void lstm_layer_kernel(const float*    __restrict__ xf,
                                  const _Float16*              xh,     // may alias hout
                                  _Float16*                    hout,   // [B,T,H] fp16
                                  const float*    __restrict__ Whh,    // [512,128]
                                  const float*    __restrict__ Wih,    // [512,KIN]
                                  const float*    __restrict__ bih,
                                  const float*    __restrict__ bhh)
{
    constexpr int XSTR = CHUNK_X ? LDX0_ : LDK_;             // x row stride (halves)
    constexpr int XBUF = CHUNK_X ? (CHX_ * BT_ * LDX0_)      // one big chunk
                                 : (2 * BT_ * LDK_);         // double-buffered tile
    extern __shared__ _Float16 smem[];
    _Float16* h_s = smem;                 // 2 * BT_*LDK_  (double-buffered h tile)
    _Float16* x_s = smem + 2 * BT_ * LDK_;

    const int tid   = threadIdx.x;
    const int wave  = tid >> 5;          // 8 waves; wave owns hidden cols 16w..16w+15
    const int lane  = tid & 31;
    const int hf    = lane >> 4;         // WMMA lane-half
    const int l16   = lane & 15;
    const int bBase = blockIdx.x * BT_;
    const int j     = wave * 16 + l16;   // hidden column this lane owns (all 4 gates)

    // ---- one-time setup: zero tiles (zero-pads persist; staging writes k<KIN only)
    for (int idx = tid; idx < 2 * BT_ * LDK_; idx += 256) h_s[idx] = (_Float16)0.0f;
    for (int idx = tid; idx < XBUF;           idx += 256) x_s[idx] = (_Float16)0.0f;

    // W_hh B-fragments -> registers (B column n = W row q*128+j, contiguous fp32).
    v16h wBh[4][4];
#pragma unroll
    for (int q = 0; q < 4; ++q)
#pragma unroll
        for (int kb = 0; kb < 4; ++kb) {
            const float* p = &Whh[(size_t)(q * H_ + j) * H_ + kb * 32 + hf * 8];
            v16h f;
#pragma unroll
            for (int i = 0; i < 16; ++i) f[i] = (_Float16)p[i + (i >= 8 ? 8 : 0)];
            wBh[q][kb] = f;
        }

    // W_ih B-fragments -> registers (zero-padded beyond KIN).
    v16h wBx[4][KXS];
#pragma unroll
    for (int q = 0; q < 4; ++q)
#pragma unroll
        for (int kb = 0; kb < KXS; ++kb) {
            v16h f;
#pragma unroll
            for (int i = 0; i < 16; ++i) {
                int k = kb * 32 + hf * 8 + i + (i >= 8 ? 8 : 0);
                f[i] = (k < KIN) ? (_Float16)Wih[(size_t)(q * H_ + j) * KIN + k]
                                 : (_Float16)0.0f;
            }
            wBx[q][kb] = f;
        }

    float bias[4];
#pragma unroll
    for (int q = 0; q < 4; ++q) bias[q] = bih[q * H_ + j] + bhh[q * H_ + j];

    float cst[8];                        // cell state, register-resident for all T
#pragma unroll
    for (int r = 0; r < 8; ++r) cst[r] = 0.0f;

    // stage x_0 (fp16 path only; chunked path stages at t==0 inside the loop)
    if constexpr (!CHUNK_X) {
        int m = tid >> 4, k = (tid & 15) * 8;
        *(v8h*)&x_s[m * XSTR + k] = *(const v8h*)&xh[((size_t)(bBase + m) * T_) * H_ + k];
    }
    __syncthreads();

    for (int t = 0; t < T_; ++t) {
        _Float16* hcur = h_s + (t & 1) * (BT_ * LDK_);
        _Float16* hnxt = h_s + ((t + 1) & 1) * (BT_ * LDK_);

        const _Float16* xrow;   // base of this lane's A-row in the x tile
        v8h pf_h = {};
        _Float16* xnxt = nullptr;

        if constexpr (CHUNK_X) {
            // ---- refill the 64-step x chunk (amortized global traffic) ----
            if ((t & (CHX_ - 1)) == 0) {
                for (int idx = tid; idx < CHX_ * BT_ * KIN; idx += 256) {
                    int tt = idx / (BT_ * KIN), rem = idx % (BT_ * KIN);
                    int m = rem / KIN, i = rem % KIN;
                    x_s[(tt * BT_ + m) * XSTR + i] =
                        (_Float16)xf[((size_t)(bBase + m) * T_ + t + tt) * KIN + i];
                }
                __syncthreads();
            }
            xrow = &x_s[((t & (CHX_ - 1)) * BT_ + l16) * XSTR];
        } else {
            _Float16* xcur = x_s + (t & 1) * (BT_ * LDK_);
            xnxt = x_s + ((t + 1) & 1) * (BT_ * LDK_);
            xrow = &xcur[l16 * LDK_];
            // ---- prefetch x_{t+1} into the other buffer; overlaps the WMMAs ----
            const int tn = (t + 1 < T_) ? t + 1 : t;
            int m = tid >> 4, k = (tid & 15) * 8;
#ifdef USE_ASYNC_LDS
            __builtin_amdgcn_global_load_async_to_lds_b128(
                (GAS_ v4i*)&xh[((size_t)(bBase + m) * T_ + tn) * H_ + k],
                (LAS_ v4i*)&xnxt[m * LDK_ + k],
                0, 0);
#else
            pf_h = *(const v8h*)&xh[((size_t)(bBase + m) * T_ + tn) * H_ + k];
#endif
        }
        (void)pf_h;

        // ---- gates: 4 independent accumulator chains, A loaded per K-step ----
        v8f acc[4] = {};
#pragma unroll
        for (int kb = 0; kb < 4; ++kb) {                 // h-path, K=128
            const _Float16* rp = &hcur[l16 * LDK_ + kb * 32 + hf * 8];
            v16h ah = cat8(*(const v8h*)rp, *(const v8h*)(rp + 16));
#pragma unroll
            for (int q = 0; q < 4; ++q)
                acc[q] = __builtin_amdgcn_wmma_f32_16x16x32_f16(false, ah, false, wBh[q][kb],
                                                                (short)0, acc[q], false, false);
        }
#pragma unroll
        for (int kb = 0; kb < KXS; ++kb) {               // x-path
            const _Float16* rp = xrow + kb * 32 + hf * 8;
            v16h ax = cat8(*(const v8h*)rp, *(const v8h*)(rp + 16));
#pragma unroll
            for (int q = 0; q < 4; ++q)
                acc[q] = __builtin_amdgcn_wmma_f32_16x16x32_f16(false, ax, false, wBx[q][kb],
                                                                (short)0, acc[q], false, false);
        }

        // ---- elementwise LSTM cell, all in registers (v_exp + v_rcp only) ----
        float hnew[8];
#pragma unroll
        for (int r = 0; r < 8; ++r) {
            float ii = sigmoid_f(acc[0][r] + bias[0]);
            float ff = sigmoid_f(acc[1][r] + bias[1]);
            float tg = tanh_f   (acc[2][r] + bias[2]);
            float oo = sigmoid_f(acc[3][r] + bias[3]);
            float cn = ff * cst[r] + ii * tg;
            cst[r]  = cn;
            hnew[r] = oo * tanh_f(cn);
        }

        // ---- land prefetch (fallback), publish h_t into the *other* h buffer ----
        if constexpr (!CHUNK_X) {
#ifndef USE_ASYNC_LDS
            int m = tid >> 4, k = (tid & 15) * 8;
            *(v8h*)&xnxt[m * LDK_ + k] = pf_h;
#endif
#ifdef USE_ASYNC_LDS
            __builtin_amdgcn_s_wait_asynccnt(0);
#endif
        }
#pragma unroll
        for (int r = 0; r < 8; ++r) {
            int m = hf * 8 + r;
            _Float16 hv = (_Float16)hnew[r];
            hnxt[m * LDK_ + j] = hv;                      // double buffer: no WAR hazard
            hout[((size_t)(bBase + m) * T_ + t) * H_ + j] = hv;
        }
        __syncthreads();   // single barrier per step: h_t and x_{t+1} now visible
    }
}

// out[b] = h2[b, T-1, :] . W_out + b_out
__global__ void lstm_head_kernel(const _Float16* __restrict__ seq,
                                 const float* __restrict__ Wout,
                                 const float* __restrict__ bout,
                                 float* __restrict__ out)
{
    int b = blockIdx.x * blockDim.x + threadIdx.x;
    if (b < B_) {
        const _Float16* hp = &seq[((size_t)b * T_ + (T_ - 1)) * H_];
        float s = bout[0];
#pragma unroll 8
        for (int k = 0; k < H_; ++k) s += (float)hp[k] * Wout[k];
        out[b] = s;
    }
}

extern "C" void kernel_launch(void* const* d_in, const int* in_sizes, int n_in,
                              void* d_out, int out_size, void* d_ws, size_t ws_size,
                              hipStream_t stream) {
    (void)in_sizes; (void)n_in; (void)out_size; (void)ws_size;
    const float* x    = (const float*)d_in[0];
    const float* Wih0 = (const float*)d_in[1];
    const float* Whh0 = (const float*)d_in[2];
    const float* bih0 = (const float*)d_in[3];
    const float* bhh0 = (const float*)d_in[4];
    const float* Wih1 = (const float*)d_in[5];
    const float* Whh1 = (const float*)d_in[6];
    const float* bih1 = (const float*)d_in[7];
    const float* bhh1 = (const float*)d_in[8];
    const float* Wih2 = (const float*)d_in[9];
    const float* Whh2 = (const float*)d_in[10];
    const float* bih2 = (const float*)d_in[11];
    const float* bhh2 = (const float*)d_in[12];
    const float* Wout = (const float*)d_in[13];
    const float* bout = (const float*)d_in[14];

    _Float16* seq = (_Float16*)d_ws;   // [B,T,H] fp16, 64 MB, reused in-place

    // LDS: 2 h-buffers + x region (layer0: 64-step chunk; layers 1/2: 2 tiles)
    const size_t lds0 = (size_t)(2 * BT_ * LDK_ + CHX_ * BT_ * LDX0_) * sizeof(_Float16);
    const size_t lds1 = (size_t)(2 * BT_ * LDK_ + 2 * BT_ * LDK_) * sizeof(_Float16);

    (void)hipFuncSetAttribute((const void*)&lstm_layer_kernel<1, true, 6>,
                              hipFuncAttributeMaxDynamicSharedMemorySize, (int)lds0);
    (void)hipFuncSetAttribute((const void*)&lstm_layer_kernel<4, false, H_>,
                              hipFuncAttributeMaxDynamicSharedMemorySize, (int)lds1);

    dim3 grid(B_ / BT_), block(256);
    lstm_layer_kernel<1, true, 6><<<grid, block, lds0, stream>>>(
        x, nullptr, seq, Whh0, Wih0, bih0, bhh0);
    lstm_layer_kernel<4, false, H_><<<grid, block, lds1, stream>>>(
        nullptr, seq, seq, Whh1, Wih1, bih1, bhh1);
    lstm_layer_kernel<4, false, H_><<<grid, block, lds1, stream>>>(
        nullptr, seq, seq, Whh2, Wih2, bih2, bhh2);
    lstm_head_kernel<<<1, 128, 0, stream>>>(seq, Wout, bout, (float*)d_out);
}
